// WindowMoBA_14276471292734
// MI455X (gfx1250) — matrix operand order repaired and checked by reading, not compiled, not run on hardware
//
#include <hip/hip_runtime.h>
#include <stdint.h>

// ---------------------------------------------------------------- constants
#define DIMC   256
#define HEADS  8
#define HD     32        // head dim
#define NTOK   49
#define NPAD   64
#define BTOT   512       // B_ = batch * num_win
#define MROWS  25088     // BTOT * NTOK
#define QK_SCALE 0.17677669529663687f   // 32^-0.5

typedef __bf16 bf16_t;
typedef __bf16  bf16x8  __attribute__((ext_vector_type(8)));
typedef __bf16  bf16x16 __attribute__((ext_vector_type(16)));
typedef float   f32x8   __attribute__((ext_vector_type(8)));
typedef unsigned int u32x4 __attribute__((ext_vector_type(4)));
typedef int     i32x4  __attribute__((ext_vector_type(4)));
typedef int     i32x8  __attribute__((ext_vector_type(8)));

// float -> bf16 round-to-nearest-even via integer math (no reliance on cast codegen)
__device__ __forceinline__ bf16_t to_bf16(float f) {
    union { float f; uint32_t u; } v; v.f = f;
    uint32_t r = v.u + 0x7FFFu + ((v.u >> 16) & 1u);
    uint16_t h = (uint16_t)(r >> 16);
    return __builtin_bit_cast(bf16_t, h);
}

__device__ __forceinline__ f32x8 wmma_bf16(bf16x16 a, bf16x16 b, f32x8 c) {
    // D = A(16x32) * B(32x16) + C, f32 accumulate
    return __builtin_amdgcn_wmma_f32_16x16x32_bf16(false, a, false, b, (short)0, c,
                                                   false, false);
}

// A-operand k order used here (16-bit 16x32): per lane two contiguous 8-elem chunks at
// K = 16*(lane/16) and K = 16*(lane/16)+8 (k = 16h+s), same order as B.
__device__ __forceinline__ bf16x16 load_a16(const bf16_t* p0, const bf16_t* p1) {
    bf16x8 lo = *(const bf16x8*)p0;
    bf16x8 hi = *(const bf16x8*)p1;
    return __builtin_shufflevector(lo, hi, 0,1,2,3,4,5,6,7,8,9,10,11,12,13,14,15);
}

// ---------------------------------------------------------------- TDM support
#if defined(__has_builtin)
#if __has_builtin(__builtin_amdgcn_tensor_load_to_lds) && \
    __has_builtin(__builtin_amdgcn_s_wait_tensorcnt)
#define HAVE_TDM 1
#endif
#endif

#ifdef HAVE_TDM
// Issue a 2D TDM tile load (bf16 elements) global -> LDS.
// D# built per CDNA5 ISA 08_async_tensor: group0 {count=1, lds_addr, global_addr,
// type=2}; group1 {data_size=1(2B), tensor_dim0/1, tile_dim0/1, dim0_stride}.
__device__ __forceinline__ void tdm_load_2d_bf16(uint32_t lds_off, const void* gptr,
                                                 uint32_t dim0, uint32_t dim1,
                                                 uint32_t tile0, uint32_t tile1,
                                                 uint32_t stride0) {
    uint64_t ga = (uint64_t)(uintptr_t)gptr;
    u32x4 g0 = { 1u,                                        // count=1, no gather
                 lds_off,                                   // lds_addr (bytes)
                 (uint32_t)ga,                              // global_addr[31:0]
                 ((uint32_t)(ga >> 32) & 0x01FFFFFFu) | (2u << 30) }; // [56:32] | type=2
    i32x8 g1 = { (int)(1u << 16),                           // data_size=1 (2 bytes)
                 (int)(dim0 << 16),                         // tensor_dim0 -> bits[63:48]
                 (int)((dim0 >> 16) | (dim1 << 16)),        // [79:64]=dim0_hi [95:80]=dim1_lo
                 (int)((dim1 >> 16) | (tile0 << 16)),       // [111:96]=dim1_hi [127:112]=tile0
                 (int)(tile1 & 0xFFFFu),                    // [143:128]=tile1, tile2=0
                 (int)stride0,                              // dim0_stride[31:0]
                 0, 0 };                                    // stride hi / dim1_stride = 0
    i32x4 z4 = { 0, 0, 0, 0 };
#if __clang_major__ >= 23
    i32x8 z8 = { 0, 0, 0, 0, 0, 0, 0, 0 };
    __builtin_amdgcn_tensor_load_to_lds(g0, g1, z4, z4, z8, 0);
#else
    __builtin_amdgcn_tensor_load_to_lds(g0, g1, z4, z4, 0);
#endif
}
#endif

// ---------------------------------------------------------------- converts
__global__ void k_convert(const float* __restrict__ in, bf16_t* __restrict__ out, int n) {
    int i = blockIdx.x * blockDim.x + threadIdx.x;
    int stride = gridDim.x * blockDim.x;
    for (; i < n; i += stride) out[i] = to_bf16(in[i]);
}

// in: (K,N) row-major f32  ->  out: (N,K) bf16
__global__ void k_transpose_convert(const float* __restrict__ in, bf16_t* __restrict__ out,
                                    int K, int N) {
    int i = blockIdx.x * blockDim.x + threadIdx.x;
    int stride = gridDim.x * blockDim.x;
    int total = K * N;
    for (; i < total; i += stride) {
        int n = i % N, k = i / N;
        out[n * K + k] = to_bf16(in[i]);
    }
}

__global__ void k_zero(uint32_t* __restrict__ p, int nwords) {
    int i = blockIdx.x * blockDim.x + threadIdx.x;
    int stride = gridDim.x * blockDim.x;
    for (; i < nwords; i += stride) p[i] = 0u;
}

// ---------------------------------------------------------------- QKV GEMM
// xb: (25088, 256) bf16 ; wT: (768, 256) bf16 (N-major) ; bias: (768) f32
// qb/kb: (512, 8, 64, 32) bf16 (token-padded, q pre-scaled) ; vt: (512, 8, 32, 64) bf16
__global__ __launch_bounds__(128)
void k_qkv(const bf16_t* __restrict__ xb, const bf16_t* __restrict__ wT,
           const float* __restrict__ bias,
           bf16_t* __restrict__ qb, bf16_t* __restrict__ kb, bf16_t* __restrict__ vt) {
    const int lane = threadIdx.x & 31;
    const int wave = threadIdx.x >> 5;
    const int half = lane >> 4;
    const int lm   = lane & 15;
    const int row0 = (blockIdx.x * 4 + wave) * 16;
    const int n0   = blockIdx.y * 64;

    f32x8 C[4] = {};
    const bf16_t* arow = xb + (size_t)(row0 + lm) * DIMC;
    for (int kk = 0; kk < 8; ++kk) {
        if (kk < 7) __builtin_prefetch(arow + (kk + 1) * 32, 0, 0);
        bf16x16 A = load_a16(arow + kk * 32 + 16 * half,
                             arow + kk * 32 + 16 * half + 8);
#pragma unroll
        for (int t = 0; t < 4; ++t) {
            bf16x16 B = *(const bf16x16*)(wT + (size_t)(n0 + t * 16 + lm) * DIMC
                                          + kk * 32 + 16 * half);
            C[t] = wmma_bf16(A, B, C[t]);
        }
    }
#pragma unroll
    for (int t = 0; t < 4; ++t) {
        int n = n0 + t * 16 + lm;
        float bv = bias[n];
        int chunk = n >> 8, rem = n & 255, h = rem >> 5, d = rem & 31;
#pragma unroll
        for (int g = 0; g < 8; ++g) {
            int row = row0 + g + 8 * half;
            float val = C[t][g] + bv;
            int b_ = row / NTOK, tok = row - b_ * NTOK;
            size_t bh = (size_t)(b_ * HEADS + h);
            if (chunk == 0)      qb[(bh * NPAD + tok) * HD + d] = to_bf16(val * QK_SCALE);
            else if (chunk == 1) kb[(bh * NPAD + tok) * HD + d] = to_bf16(val);
            else                 vt[(bh * HD + d) * NPAD + tok] = to_bf16(val);
        }
    }
}

// ---------------------------------------------------------------- attention
// One wave per (m, h). Accumulates all 4 t_out terms into the same C tiles.
// K/V tiles are t_out-invariant: staged to LDS via the Tensor Data Mover.
__global__ __launch_bounds__(32)
void k_attn(const bf16_t* __restrict__ qb, const bf16_t* __restrict__ kb,
            const bf16_t* __restrict__ vt, const float* __restrict__ relt,
            bf16_t* __restrict__ avg) {
    __shared__ float  S[NPAD * NPAD];
    __shared__ bf16_t P[NPAD * NPAD];
    __shared__ bf16_t KT[NPAD * HD];   // (token, d)
    __shared__ bf16_t VT[HD * NPAD];   // (d, token)
    __shared__ float  biasL[169];

    const int m = blockIdx.x, h = blockIdx.y;
    const int lane = threadIdx.x, half = lane >> 4, lm = lane & 15;

    const int kv = ((m >> 3) & 3) * 8 + (m & 7);            // K/V window*batch index
    const bf16_t* kp = kb + (size_t)(kv * HEADS + h) * NPAD * HD;
    const bf16_t* vp = vt + (size_t)(kv * HEADS + h) * HD * NPAD;

#ifdef HAVE_TDM
    // Async DMA of the 64x32 K tile and 32x64 V^T tile into LDS (4 KB each).
    tdm_load_2d_bf16((uint32_t)(uintptr_t)KT, kp, HD, NPAD, HD, NPAD, HD);
    tdm_load_2d_bf16((uint32_t)(uintptr_t)VT, vp, NPAD, HD, NPAD, HD, NPAD);
#endif

    // overlap: per-head rel-pos bias table into LDS
    for (int idx = lane; idx < 169; idx += 32) biasL[idx] = relt[idx * HEADS + h];

#ifdef HAVE_TDM
    __builtin_amdgcn_s_wait_tensorcnt(0);
    const bf16_t* kpl = KT;
    const bf16_t* vpl = VT;
#else
    const bf16_t* kpl = kp;
    const bf16_t* vpl = vp;
#endif
    __syncthreads();

    // B operands are invariant across t_out: hoist into registers (from LDS).
    bf16x16 Bk[4];
#pragma unroll
    for (int tn = 0; tn < 4; ++tn)
        Bk[tn] = *(const bf16x16*)(kpl + (tn * 16 + lm) * HD + 16 * half);
    bf16x16 Bv[2][2];
#pragma unroll
    for (int s = 0; s < 2; ++s)
#pragma unroll
        for (int tn = 0; tn < 2; ++tn)
            Bv[s][tn] = *(const bf16x16*)(vpl + (tn * 16 + lm) * NPAD + s * 32 + 16 * half);

    f32x8 O[4][2] = {};
    __syncthreads();

    for (int to = 0; to < 4; ++to) {
        const int qi = to * 128 + (m >> 2);
        const bf16_t* qp = qb + (size_t)(qi * HEADS + h) * NPAD * HD;

        // S = (q*scale) @ K^T   (K dim = 32 = one WMMA)
        for (int tm = 0; tm < 4; ++tm) {
            const bf16_t* ar = qp + (tm * 16 + lm) * HD;
            bf16x16 A = load_a16(ar + 16 * half, ar + 16 * half + 8);
#pragma unroll
            for (int tn = 0; tn < 4; ++tn) {
                f32x8 z = {};
                f32x8 Sr = wmma_bf16(A, Bk[tn], z);
#pragma unroll
                for (int g = 0; g < 8; ++g)
                    S[(tm * 16 + g + 8 * half) * NPAD + tn * 16 + lm] = Sr[g];
            }
        }
        __syncthreads();

        // softmax(S + rel_bias) rows: lane handles rows lane and lane+32
#pragma unroll
        for (int rsel = 0; rsel < 2; ++rsel) {
            int rr = lane + rsel * 32;
            if (rr < NTOK) {
                int qy = rr / 7, qx = rr - qy * 7;
                float mx = -1e30f;
                int ky = 0, kx = 0;
                for (int j = 0; j < NTOK; ++j) {
                    float sv = S[rr * NPAD + j]
                             + biasL[(qy - ky + 6) * 13 + (qx - kx + 6)];
                    S[rr * NPAD + j] = sv;
                    mx = fmaxf(mx, sv);
                    if (++kx == 7) { kx = 0; ++ky; }
                }
                float sum = 0.f;
                for (int j = 0; j < NTOK; ++j) {
                    float e = __expf(S[rr * NPAD + j] - mx);
                    sum += e;
                    S[rr * NPAD + j] = e;
                }
                float inv = 1.0f / sum;
                for (int j = 0; j < NTOK; ++j) P[rr * NPAD + j] = to_bf16(S[rr * NPAD + j] * inv);
                for (int j = NTOK; j < NPAD; ++j) P[rr * NPAD + j] = to_bf16(0.f);
            } else if (rr < NPAD) {
                for (int j = 0; j < NPAD; ++j) P[rr * NPAD + j] = to_bf16(0.f);
            }
        }
        __syncthreads();

        // O += P @ V   (64 keys = 2 WMMA K-steps)
#pragma unroll
        for (int s = 0; s < 2; ++s) {
#pragma unroll
            for (int tm = 0; tm < 4; ++tm) {
                const bf16_t* pr = &P[(tm * 16 + lm) * NPAD + s * 32 + 16 * half];
                bf16x16 A = load_a16(pr, pr + 8);
#pragma unroll
                for (int tn = 0; tn < 2; ++tn)
                    O[tm][tn] = wmma_bf16(A, Bv[s][tn], O[tm][tn]);
            }
        }
        __syncthreads();   // S reused next t_out
    }

    // mean over t_out, store bf16 into (m, tok_pad64, h*32+d)
#pragma unroll
    for (int tm = 0; tm < 4; ++tm)
#pragma unroll
        for (int tn = 0; tn < 2; ++tn)
#pragma unroll
            for (int g = 0; g < 8; ++g) {
                int tok = tm * 16 + g + 8 * half;
                if (tok < NTOK)
                    avg[((size_t)m * NPAD + tok) * DIMC + h * HD + tn * 16 + lm]
                        = to_bf16(0.25f * O[tm][tn][g]);
            }
}

// ---------------------------------------------------------------- out proj
// ab: (512*64, 256) bf16 (token-padded rows; pad rows garbage, discarded)
// wT: (256, 256) bf16 (N-major) ; out: (25088, 256) f32
__global__ __launch_bounds__(128)
void k_proj(const bf16_t* __restrict__ ab, const bf16_t* __restrict__ wT,
            const float* __restrict__ bias, float* __restrict__ out) {
    const int lane = threadIdx.x & 31;
    const int wave = threadIdx.x >> 5;
    const int half = lane >> 4;
    const int lm   = lane & 15;
    const int row0 = (blockIdx.x * 4 + wave) * 16;
    const int n0   = blockIdx.y * 64;

    f32x8 C[4] = {};
    const bf16_t* arow = ab + (size_t)(row0 + lm) * DIMC;
    for (int kk = 0; kk < 8; ++kk) {
        if (kk < 7) __builtin_prefetch(arow + (kk + 1) * 32, 0, 0);
        bf16x16 A = load_a16(arow + kk * 32 + 16 * half,
                             arow + kk * 32 + 16 * half + 8);
#pragma unroll
        for (int t = 0; t < 4; ++t) {
            bf16x16 B = *(const bf16x16*)(wT + (size_t)(n0 + t * 16 + lm) * DIMC
                                          + kk * 32 + 16 * half);
            C[t] = wmma_bf16(A, B, C[t]);
        }
    }
#pragma unroll
    for (int t = 0; t < 4; ++t) {
        int n = n0 + t * 16 + lm;
        float bv = bias[n];
#pragma unroll
        for (int g = 0; g < 8; ++g) {
            int row = row0 + g + 8 * half;
            int mm = row >> 6, tok = row & 63;
            if (tok < NTOK)
                out[((size_t)mm * NTOK + tok) * DIMC + n] = C[t][g] + bv;
        }
    }
}

// ---------------------------------------------------------------- launch
extern "C" void kernel_launch(void* const* d_in, const int* in_sizes, int n_in,
                              void* d_out, int out_size, void* d_ws, size_t ws_size,
                              hipStream_t stream) {
    const float* x      = (const float*)d_in[0];
    const float* w_qkv  = (const float*)d_in[1];
    const float* b_qkv  = (const float*)d_in[2];
    const float* relt   = (const float*)d_in[3];
    const float* w_proj = (const float*)d_in[4];
    const float* b_proj = (const float*)d_in[5];
    (void)in_sizes; (void)n_in; (void)out_size; (void)ws_size;

    char* ws = (char*)d_ws;
    size_t off = 0;
    bf16_t* xb     = (bf16_t*)(ws + off); off += (size_t)MROWS * DIMC * 2;          // 12.25 MB
    bf16_t* wqkvT  = (bf16_t*)(ws + off); off += (size_t)768 * 256 * 2;
    bf16_t* wprojT = (bf16_t*)(ws + off); off += (size_t)256 * 256 * 2;
    bf16_t* qb     = (bf16_t*)(ws + off); off += (size_t)BTOT * HEADS * NPAD * HD * 2;
    bf16_t* kb     = (bf16_t*)(ws + off); off += (size_t)BTOT * HEADS * NPAD * HD * 2;
    bf16_t* vt     = (bf16_t*)(ws + off); off += (size_t)BTOT * HEADS * HD * NPAD * 2;
    bf16_t* avg    = (bf16_t*)(ws + off); off += (size_t)BTOT * NPAD * DIMC * 2;

    k_convert<<<4096, 256, 0, stream>>>(x, xb, MROWS * DIMC);
    k_transpose_convert<<<768, 256, 0, stream>>>(w_qkv, wqkvT, 256, 768);
    k_transpose_convert<<<256, 256, 0, stream>>>(w_proj, wprojT, 256, 256);
    // zero q/k/vt (contiguous) so token padding rows 49..63 are exact zeros
    k_zero<<<4096, 256, 0, stream>>>((uint32_t*)qb,
                                     (int)((size_t)3 * BTOT * HEADS * NPAD * HD * 2 / 4));

    k_qkv<<<dim3(392, 12), 128, 0, stream>>>(xb, wqkvT, b_qkv, qb, kb, vt);
    k_attn<<<dim3(512, 8), 32, 0, stream>>>(qb, kb, vt, relt, avg);
    k_proj<<<dim3(512, 4), 128, 0, stream>>>(avg, wprojT, b_proj, (float*)d_out);
}
